// MetaBaseline_23510650978982
// MI455X (gfx1250) — matrix operand order repaired and checked by reading, not compiled, not run on hardware
//
#include <hip/hip_runtime.h>
#include <hip/hip_bf16.h>

#define Bn    2
#define Qn    75
#define WAYn  5
#define SHOTn 5
#define HWn   100
#define Cn    640
#define NK    5

#define MP    112              // padded M rows per (b,q)   (7 x 16)
#define NP    512              // padded N cols per (b,way) (32 x 16)

typedef __attribute__((ext_vector_type(16))) __bf16       v16bf;
typedef __attribute__((ext_vector_type(8)))  __bf16       v8bf;
typedef __attribute__((ext_vector_type(8)))  float        v8f;
typedef __attribute__((ext_vector_type(4)))  unsigned int v4u;

union ABFrag { v16bf bf; v8bf h[2]; };

static __device__ __forceinline__ unsigned int f2bf(float f) {
  unsigned int u = __float_as_uint(f);
  u += 0x7FFFu + ((u >> 16) & 1u);        // round-to-nearest-even
  return (u >> 16) & 0xFFFFu;
}

// ---------------------------------------------------------------------------
// Kernel 0: zero the padded bf16 workspace (16 B per thread).
// ---------------------------------------------------------------------------
__global__ void zero_ws_kernel(v4u* __restrict__ p) {
  p[blockIdx.x * blockDim.x + threadIdx.x] = (v4u)0u;
}

// ---------------------------------------------------------------------------
// Kernel 1a: L2-normalize query descriptors -> padded [B*Q][112][640] bf16.
// One wave32 per descriptor; lane L owns channel pairs p = L + 32*i, i<10.
// ---------------------------------------------------------------------------
__global__ void normalize_q_kernel(const float* __restrict__ in,
                                   unsigned int* __restrict__ out) {
  const int lane = threadIdx.x & 31;
  const int wv   = threadIdx.x >> 5;
  const int desc = blockIdx.x * (blockDim.x >> 5) + wv;   // [0, 15000)

  const float* p = in + (size_t)desc * Cn;
  float2 d[10];
  float ss = 0.f;
#pragma unroll
  for (int i = 0; i < 10; ++i) {
    d[i] = *(const float2*)(p + 2 * (lane + 32 * i));
    ss += d[i].x * d[i].x + d[i].y * d[i].y;
  }
#pragma unroll
  for (int off = 16; off >= 1; off >>= 1) ss += __shfl_xor(ss, off, 32);
  const float inv = rsqrtf(ss);

  const int drow = (desc / HWn) * MP + (desc % HWn);      // padded row
  unsigned int* o = out + (size_t)drow * (Cn / 2);
#pragma unroll
  for (int i = 0; i < 10; ++i) {
    unsigned int lo = f2bf(d[i].x * inv);
    unsigned int hi = f2bf(d[i].y * inv);
    o[lane + 32 * i] = lo | (hi << 16);
  }
}

// ---------------------------------------------------------------------------
// Kernel 1b: L2-normalize support descriptors -> padded, regrouped
// [B*WAY][512][640] bf16.  Reference ".view" regroup: for flat desc d,
// class k = (d % 2500) / 500, column n = d % 500.
// ---------------------------------------------------------------------------
__global__ void normalize_s_kernel(const float* __restrict__ in,
                                   unsigned int* __restrict__ out) {
  const int lane = threadIdx.x & 31;
  const int wv   = threadIdx.x >> 5;
  const int desc = blockIdx.x * (blockDim.x >> 5) + wv;   // [0, 5000)

  const float* p = in + (size_t)desc * Cn;
  float2 d[10];
  float ss = 0.f;
#pragma unroll
  for (int i = 0; i < 10; ++i) {
    d[i] = *(const float2*)(p + 2 * (lane + 32 * i));
    ss += d[i].x * d[i].x + d[i].y * d[i].y;
  }
#pragma unroll
  for (int off = 16; off >= 1; off >>= 1) ss += __shfl_xor(ss, off, 32);
  const float inv = rsqrtf(ss);

  const int b    = desc / (SHOTn * WAYn * HWn);
  const int rem  = desc % (SHOTn * WAYn * HWn);
  const int k    = rem / (WAYn * HWn);                    // class (= shot idx)
  const int n    = rem % (WAYn * HWn);                    // column within class
  const int drow = (b * WAYn + k) * NP + n;
  unsigned int* o = out + (size_t)drow * (Cn / 2);
#pragma unroll
  for (int i = 0; i < 10; ++i) {
    unsigned int lo = f2bf(d[i].x * inv);
    unsigned int hi = f2bf(d[i].y * inv);
    o[lane + 32 * i] = lo | (hi << 16);
  }
}

// ---------------------------------------------------------------------------
// Kernel 2: per (b,q,way): 112x512 (padded) cosine-score GEMM via bf16 WMMA,
// running top-5 per valid row, mean over 5, sum over valid rows.
// Block = 256 threads = 8 waves; wave w owns N-tile w of each 128-col chunk
// and all 7 M-tiles (56 accumulator VGPRs live across K=640).
// A-fragments use a 2-deep ping-pong (a0/a1): the loads for M-tile m+1 are
// issued before the WMMA consuming tile m, so each WMMA waits on loads issued
// one stage earlier (s_wait_loadcnt 2) instead of fully serializing, while
// keeping register pressure low (no spills).
// ---------------------------------------------------------------------------
__global__ void __launch_bounds__(256)
scores_topk_kernel(const __bf16* __restrict__ Aq,   // [B*Q][MP][C] padded bf16
                   const __bf16* __restrict__ Bs,   // [B*WAY][NP][C] padded bf16
                   float* __restrict__ out) {
  __shared__ float sc[MP * 129];    // score chunk, padded stride (57792 B)
  __shared__ float red[256];

  const int bid = blockIdx.x;
  const int way = bid % WAYn;
  const int q   = (bid / WAYn) % Qn;
  const int b   = bid / (WAYn * Qn);

  const int t    = threadIdx.x;
  const int lane = t & 31;
  const int wv   = t >> 5;
  const int half = lane >> 4;     // 0: low half-wave, 1: high half-wave
  const int l16  = lane & 15;

  const __bf16* Abase = Aq + (size_t)(b * Qn + q) * MP * Cn;
  const __bf16* Bbase = Bs + (size_t)(b * WAYn + way) * NP * Cn;

  // A fragment base for this lane (16-bit A 16x32 layout:
  //  lane<16 -> K=k0+0..7 & k0+16..23, lane>=16 -> K=k0+8..15 & k0+24..31,
  //  row = lane%16 within the M-tile).
  const __bf16* Alane = Abase + (size_t)l16 * Cn + half * 8;

  // running top-5 (descending) for row t
  float t0 = -1e30f, t1 = -1e30f, t2 = -1e30f, t3 = -1e30f, t4 = -1e30f;

  for (int nc = 0; nc < 4; ++nc) {
    // B fragment: column n = one support descriptor (contiguous in memory).
    // 16-bit B 32x16 layout: lane<16 -> K=k0+0..7 & k0+16..23, lane>=16 -> K=k0+8..15 & k0+24..31.
    const int n = (nc * 8 + wv) * 16 + l16;               // column 0..511
    const __bf16* Blane = Bbase + (size_t)n * Cn + half * 8;

    v8f acc[7];
#pragma unroll
    for (int m = 0; m < 7; ++m) acc[m] = (v8f)0.0f;

#pragma unroll 2
    for (int ks = 0; ks < 20; ++ks) {
      const int k0 = ks * 32;
      const __bf16* al = Alane + k0;

      ABFrag bfr;
      bfr.h[0] = *(const v8bf*)(Blane + k0);
      bfr.h[1] = *(const v8bf*)(Blane + k0 + 16);

#define LOADA(f, m)                                                  \
      do {                                                           \
        (f).h[0] = *(const v8bf*)(al + (size_t)(m) * 16 * Cn);       \
        (f).h[1] = *(const v8bf*)(al + (size_t)(m) * 16 * Cn + 16);  \
      } while (0)
#define WMMA(m, f)                                                   \
      acc[m] = __builtin_amdgcn_wmma_f32_16x16x32_bf16(              \
          false, (f).bf, false, bfr.bf, (short)0, acc[m], false, false)

      ABFrag a0, a1;
      LOADA(a0, 0);
      LOADA(a1, 1);  WMMA(0, a0);
      LOADA(a0, 2);  WMMA(1, a1);
      LOADA(a1, 3);  WMMA(2, a0);
      LOADA(a0, 4);  WMMA(3, a1);
      LOADA(a1, 5);  WMMA(4, a0);
      LOADA(a0, 6);  WMMA(5, a1);
                     WMMA(6, a0);
#undef LOADA
#undef WMMA
    }

    // C/D layout: VGPR j, lanes 0-15 -> M=j, N=lane; lanes 16-31 -> M=j+8.
    const int cl = wv * 16 + l16;    // chunk-local column
#pragma unroll
    for (int m = 0; m < 7; ++m) {
#pragma unroll
      for (int j = 0; j < 8; ++j) {
        const int row = m * 16 + j + half * 8;
        sc[row * 129 + cl] = acc[m][j];
      }
    }
    __syncthreads();

    // top-5 update: thread t scans row t of this 128-wide chunk
    if (t < HWn) {
      const int maxj = (nc < 3) ? 128 : (WAYn * HWn - 3 * 128);  // skip padded cols
      const float* rp = &sc[t * 129];
      for (int j = 0; j < maxj; ++j) {
        const float v = rp[j];
        if (v > t4) {
          if (v > t3) { t4 = t3;
            if (v > t2) { t3 = t2;
              if (v > t1) { t2 = t1;
                if (v > t0) { t1 = t0; t0 = v; } else t1 = v;
              } else t2 = v;
            } else t3 = v;
          } else t4 = v;
        }
      }
    }
    __syncthreads();
  }

  red[t] = (t < HWn) ? (t0 + t1 + t2 + t3 + t4) : 0.f;
  __syncthreads();
  if (t == 0) {
    float tot = 0.f;
    for (int i = 0; i < HWn; ++i) tot += red[i];
    out[bid] = tot * (1.0f / NK);
  }
}

// ---------------------------------------------------------------------------
extern "C" void kernel_launch(void* const* d_in, const int* in_sizes, int n_in,
                              void* d_out, int out_size, void* d_ws, size_t ws_size,
                              hipStream_t stream) {
  (void)in_sizes; (void)n_in; (void)out_size; (void)ws_size;

  const float* q_in = (const float*)d_in[0];              // [2,75,10,10,640]
  const float* s_in = (const float*)d_in[1];              // [2,5,5,10,10,640]
  // d_in[2] = neighbor_k (compile-time 5 here)

  const size_t qpad_elems = (size_t)Bn * Qn * MP * Cn;    // 10,752,000 bf16
  const size_t spad_elems = (size_t)Bn * WAYn * NP * Cn;  //  3,276,800 bf16

  __bf16* qn = (__bf16*)d_ws;                             // 21.50 MB
  __bf16* sn = qn + qpad_elems;                           //  6.55 MB

  // zero all padded storage (one 16B store per thread)
  const size_t total_bytes = (qpad_elems + spad_elems) * sizeof(__bf16); // 28,057,600
  const int    zthreads    = (int)(total_bytes / 16);                    // 1,753,600
  zero_ws_kernel<<<zthreads / 256, 256, 0, stream>>>((v4u*)d_ws);

  normalize_q_kernel<<<(Bn * Qn * HWn) / 8, 256, 0, stream>>>(
      q_in, (unsigned int*)qn);
  normalize_s_kernel<<<(Bn * SHOTn * WAYn * HWn) / 8, 256, 0, stream>>>(
      s_in, (unsigned int*)sn);

  scores_topk_kernel<<<Bn * Qn * WAYn, 256, 0, stream>>>(
      qn, sn, (float*)d_out);
}
